// Global_Filter_78417512890601
// MI455X (gfx1250) — matrix operand order repaired and checked by reading, not compiled, not run on hardware
//
#include <hip/hip_runtime.h>

typedef __attribute__((ext_vector_type(16))) _Float16 v16h;
typedef __attribute__((ext_vector_type(8)))  float    v8f;
typedef __attribute__((ext_vector_type(4)))  unsigned int u32x4;
typedef __attribute__((ext_vector_type(4)))  int          i32x4;
typedef __attribute__((ext_vector_type(8)))  int          i32x8;

#define PI_F 3.14159265358979323846f

__device__ __forceinline__ unsigned pack_h2(_Float16 lo, _Float16 hi) {
  unsigned short a = __builtin_bit_cast(unsigned short, lo);
  unsigned short b = __builtin_bit_cast(unsigned short, hi);
  return (unsigned)a | ((unsigned)b << 16);
}
__device__ __forceinline__ _Float16 h_from(unsigned u, int sh) {
  return __builtin_bit_cast(_Float16, (unsigned short)(u >> sh));
}
__device__ __forceinline__ v8f wmma16(v16h a, v16h b, v8f c) {
  // D = A(16x32 f16) x B(32x16 f16) + C(16x16 f32)
  return __builtin_amdgcn_wmma_f32_16x16x32_f16(false, a, false, b, (short)0, c, false, false);
}

// TDM descriptor for a 2D tile: 256 rows (tokens, stride 768 f32) x 16 f32 (channels),
// stored contiguously row-major in LDS. tensor dims == tile dims -> no OOB path.
__device__ __forceinline__ void tdm_build(u32x4& g0, i32x8& g1,
                                          unsigned lds_off, unsigned long long ga) {
  g0[0] = 1u;                                             // count=1 (valid), user mode
  g0[1] = lds_off;                                        // lds_addr (bytes)
  g0[2] = (unsigned)ga;                                   // global_addr[31:0]
  g0[3] = (unsigned)((ga >> 32) & 0x01FFFFFFu) | (2u << 30); // addr[56:32] | type=2
  g1[0] = (int)(2u << 16);     // workgroup_mask=0, data_size=2 (4B), no flags
  g1[1] = (int)(16u << 16);    // atomic_barrier=0 | tensor_dim0=16 (lo16)
  g1[2] = (int)(256u << 16);   // tensor_dim0 hi=0 | tensor_dim1=256 (lo16)
  g1[3] = (int)(16u << 16);    // tensor_dim1 hi=0 | tile_dim0=16
  g1[4] = 256;                 // tile_dim1=256, tile_dim2=0
  g1[5] = 768;                 // tensor_dim0_stride[31:0]=768
  g1[6] = 0;                   // stride0 hi | stride1 lo
  g1[7] = 0;
}
__device__ __forceinline__ void tdm_load(unsigned lds_off, const void* gp) {
  u32x4 g0; i32x8 g1;
  tdm_build(g0, g1, lds_off, (unsigned long long)(uintptr_t)gp);
  i32x4 z4 = {};
#if __clang_major__ >= 23
  i32x8 z8 = {};
  __builtin_amdgcn_tensor_load_to_lds(g0, g1, z4, z4, z8, 0);
#else
  __builtin_amdgcn_tensor_load_to_lds(g0, g1, z4, z4, 0);
#endif
}
__device__ __forceinline__ void tdm_store(unsigned lds_off, void* gp) {
  u32x4 g0; i32x8 g1;
  tdm_build(g0, g1, lds_off, (unsigned long long)(uintptr_t)gp);
  i32x4 z4 = {};
#if __clang_major__ >= 23
  i32x8 z8 = {};
  __builtin_amdgcn_tensor_store_from_lds(g0, g1, z4, z4, z8, 0);
#else
  __builtin_amdgcn_tensor_store_from_lds(g0, g1, z4, z4, 0);
#endif
}

// One workgroup: batch b, channels [c0, c0+64). 4 fully independent waves, each
// owning 16 channels and a private 16KB LDS slice DMA'd in/out by the TDM.
//   stage1: Y = A*X        (along n1; K-halves carry x_hi/x_lo split, f32 X from TDM)
//   stage2: Z = A*Y        (along n2; complex pair fused into K=32)
//   weight: M = W .* Z     (Hermitian-extended weight, f32 VALU, L2-resident)
//   stage3: U = invA*M     (along k2; operands via half-wave shfl_xor, no LDS)
//   stage4: out = Re(...)  (along k1; single WMMA per batch index)
__global__ __launch_bounds__(128) void gfnet_wmma_kernel(
    const float* __restrict__ x, const float2* __restrict__ wgt,
    float* __restrict__ out)
{
  __shared__ unsigned sbuf[4 * 4096];  // 4 x 16KB slices

  const int b   = blockIdx.y;
  const int c0  = blockIdx.x * 64;
  const int tid = threadIdx.x;
  const int lane = tid & 31;
  const int wv   = tid >> 5;
  const int r    = lane & 15;  // row (A) / column (B,C,D) owned by this lane
  const int h    = lane >> 4;  // lane half

  unsigned* S = &sbuf[wv * 4096];
  const unsigned lds_off = (unsigned)(uintptr_t)S;      // LDS byte offset (addr[31:0])
  const size_t tile_base = (size_t)b * 256 * 768 + c0 + wv * 16;

  // kick off async tile DMA: global (256x16 f32, row stride 768) -> LDS slice
  tdm_load(lds_off, x + tile_base);

  // ---- per-lane DFT matrix constants while the DMA is in flight ----
  float cv[16], sv[16];
#pragma unroll
  for (int kk = 0; kk < 16; ++kk) {
    int k = kk;
    int m = (r * k) & 15;                 // exact period-16 reduction
    float th = (PI_F / 8.0f) * (float)m;  // 2*pi*r*k/16
    cv[kk] = 0.25f * __cosf(th);
    sv[kk] = 0.25f * __sinf(th);
  }
  v16h aCC, aSSn, aCS, aCmS;
#pragma unroll
  for (int e = 0; e < 16; ++e) {
    _Float16 cc = (_Float16)cv[e];
    _Float16 ss = (_Float16)sv[e];
    _Float16 ns = (_Float16)(-sv[e]);
    aCC[e]  = cc;   // [C | C ]  (hi/lo input split)
    aSSn[e] = ns;   // [-S | -S]
    aCS[e]  = h ? ss : cc;   // [C | S ]
    aCmS[e] = h ? ns : cc;   // [C | -S]
  }

  __builtin_amdgcn_s_wait_tensorcnt(0);     // tile resident in LDS
  asm volatile("" ::: "memory");            // TDM wrote LDS: no load folding

  v8f zero = {};

  // ---- stage 1: forward DFT along n1, batched over n2 (reads f32 X) ----
  const float* Xf = (const float*)S;
  for (int n2 = 0; n2 < 16; ++n2) {
    v16h bx;
#pragma unroll
    for (int e = 0; e < 16; ++e) {
      float u = Xf[(e * 16 + n2) * 16 + r];   // X[n1=e][n2][col]
      _Float16 hi = (_Float16)u;
      _Float16 lo = (_Float16)(u - (float)hi);
      bx[e] = h ? lo : hi;                    // lanes<16: x_hi, lanes>=16: x_lo
    }
    v8f yr = wmma16(aCC,  bx, zero);          // C*(x_hi+x_lo)
    v8f yi = wmma16(aSSn, bx, zero);          // -S*(x_hi+x_lo)
#pragma unroll
    for (int j = 0; j < 8; ++j) {
      int row = j + 8 * h;                    // k1
      S[(row * 16 + n2) * 16 + r] = pack_h2((_Float16)yr[j], (_Float16)yi[j]);
    }
  }

  // ---- stage 2 (forward along n2) + weight + stage 3 (inverse along k2) ----
  const int cg = c0 + wv * 16 + r;            // global channel
  for (int k1 = 0; k1 < 16; ++k1) {
    v16h bzr, bzi;
#pragma unroll
    for (int e = 0; e < 16; ++e) {
      unsigned u = S[(k1 * 16 + e) * 16 + r]; // Y[k1][n2=e][col], (Yr,Yi) packed
      bzr[e] = h_from(u, 16 * h);             // [Yr ; Yi]
      bzi[e] = h_from(u, 16 * (1 - h));       // [Yi ; Yr]
    }
    v8f zr = wmma16(aCS,  bzr, zero);         // C*Yr + S*Yi
    v8f zi = wmma16(aCmS, bzi, zero);         // C*Yi - S*Yr

    float Mr[8], Mi[8];
#pragma unroll
    for (int j = 0; j < 8; ++j) {
      int k2 = j + 8 * h;
      bool cj = (k2 > 8);                     // Hermitian-extended half
      int kk1 = cj ? ((16 - k1) & 15) : k1;
      int kk2 = cj ? (16 - k2) : k2;
      float2 wvv = wgt[(size_t)(kk1 * 9 + kk2) * 768 + cg];
      float wr = wvv.x;
      float wi = cj ? -wvv.y : wvv.y;
      Mr[j] = wr * zr[j] - wi * zi[j];
      Mi[j] = wr * zi[j] + wi * zr[j];
    }

    float sMr[8], sMi[8];
#pragma unroll
    for (int j = 0; j < 8; ++j) {             // swap wave halves: other 8 k2 rows
      sMr[j] = __shfl_xor(Mr[j], 16, 32);
      sMi[j] = __shfl_xor(Mi[j], 16, 32);
    }
    v16h b3r, b3i;                            // b3r = [Mr;Mi], b3i = [Mi;Mr]
#pragma unroll
    for (int e = 0; e < 8; ++e) {
      b3r[e]     = (_Float16)(h ? sMi[e] : Mr[e]);
      b3i[e]     = (_Float16)(h ? sMr[e] : Mi[e]);
      b3r[e + 8] = (_Float16)(h ? Mi[e]  : sMr[e]);
      b3i[e + 8] = (_Float16)(h ? Mr[e]  : sMi[e]);
    }
    v8f ur = wmma16(aCmS, b3r, zero);         // C*Mr - S*Mi
    v8f ui = wmma16(aCS,  b3i, zero);         // C*Mi + S*Mr
#pragma unroll
    for (int j = 0; j < 8; ++j) {
      int row = j + 8 * h;                    // n2
      S[(k1 * 16 + row) * 16 + r] = pack_h2((_Float16)ur[j], (_Float16)ui[j]);
    }
  }

  // ---- stage 4: inverse along k1, real output only ----
  for (int n2 = 0; n2 < 16; ++n2) {
    v16h b4;
#pragma unroll
    for (int e = 0; e < 16; ++e) {
      unsigned u = S[(e * 16 + n2) * 16 + r]; // U[k1=e][n2][col]
      b4[e] = h_from(u, 16 * h);              // [Ur ; Ui]
    }
    v8f o = wmma16(aCmS, b4, zero);           // C*Ur - S*Ui
    float* Sf = (float*)S;
#pragma unroll
    for (int j = 0; j < 8; ++j) {
      int n1 = j + 8 * h;
      Sf[(n1 * 16 + n2) * 16 + r] = o[j];
    }
  }

  // drain this wave's LDS stores, then DMA the f32 tile back out
  asm volatile("s_wait_dscnt 0" ::: "memory");
  tdm_store(lds_off, out + tile_base);
  __builtin_amdgcn_s_wait_tensorcnt(0);       // LDS must stay live until TDM done
}

extern "C" void kernel_launch(void* const* d_in, const int* in_sizes, int n_in,
                              void* d_out, int out_size, void* d_ws, size_t ws_size,
                              hipStream_t stream) {
  const float*  x = (const float*)d_in[0];
  const float2* w = (const float2*)d_in[1];
  float* o = (float*)d_out;
  dim3 grid(12, 256);   // 768/64 channel groups x 256 batches
  gfnet_wmma_kernel<<<grid, 128, 0, stream>>>(x, w, o);
}
